// MultiHeadAttention_86277303042754
// MI455X (gfx1250) — matrix operand order repaired and checked by reading, not compiled, not run on hardware
//
#include <hip/hip_runtime.h>
#include <stdint.h>

// MI455X / gfx1250: wave32, WMMA f16 16x16x32 -> f32 accum.
// Shapes fixed by the reference: B=4, S=2048, D=1024, H=16, d=64.

typedef _Float16 v16h __attribute__((ext_vector_type(16)));
typedef _Float16 v8h  __attribute__((ext_vector_type(8)));
typedef _Float16 h2   __attribute__((ext_vector_type(2)));
typedef float    v8f  __attribute__((ext_vector_type(8)));

#define L2E 1.44269504088896340736f

__device__ __forceinline__ v16h cat16(v8h lo, v8h hi) {
  return __builtin_shufflevector(lo, hi, 0,1,2,3,4,5,6,7,8,9,10,11,12,13,14,15);
}

// A fragment (16x32, row-major source): lane t holds row m=t%16;
// halves 0..7 = K[8h .. 8h+7], halves 8..15 = K[16+8h .. 16+8h+7]  (h = t/16)
__device__ __forceinline__ v16h loadA(const _Float16* p /*row base + kchunk*/, int lh) {
  v8h lo = *(const v8h*)(p + 8*lh);
  v8h hi = *(const v8h*)(p + 16 + 8*lh);
  return cat16(lo, hi);
}
// B fragment (32x16): lane t holds column n=t%16; halves 0..7 = B[8h+j][n], 8..15 = B[16+8h+(j-8)][n].
// Source stored so that "column n" is a contiguous row -> two contiguous 8-half runs per lane.
__device__ __forceinline__ v16h loadB(const _Float16* p /*row n base + kchunk*/, int lh) {
  v8h lo = *(const v8h*)(p + 8*lh);
  v8h hi = *(const v8h*)(p + 16 + 8*lh);
  return cat16(lo, hi);
}

__device__ __forceinline__ v8f wmma16(v16h a, v16h b, v8f c) {
  return __builtin_amdgcn_wmma_f32_16x16x32_f16(false, a, false, b, (short)0, c, false, false);
}

// CDNA5 async global->LDS copy (16B per lane), tracked with ASYNCcnt.
__device__ __forceinline__ void async16(_Float16* dst_lds, const _Float16* src_glb) {
  uint32_t l = (uint32_t)(uintptr_t)dst_lds;         // low 32 bits of flat addr == LDS offset
  uint64_t g = (uint64_t)(uintptr_t)src_glb;
  asm volatile("global_load_async_to_lds_b128 %0, %1, off" :: "v"(l), "v"(g) : "memory");
}
__device__ __forceinline__ void wait_async0() {
  asm volatile("s_wait_asynccnt 0x0" ::: "memory");
}

// ---------------- Kernel 1: Q = (x_h @ W + b) * 0.125, stored f16 in workspace ------
// The 1/8 on both Q and K folds the reference's 1/d (=1/64) score scale into the GEMM.
__global__ __launch_bounds__(256) void proj_kernel(const float* __restrict__ x,
                                                   const float* __restrict__ W,
                                                   const float* __restrict__ bias,
                                                   _Float16* __restrict__ Qh) {
  constexpr int S = 2048, D = 1024, d = 64;
  __shared__ __align__(16) _Float16 Xs[128 * 72];   // 128 rows x 64, padded
  __shared__ __align__(16) _Float16 WsT[64 * 72];   // W transposed: row = out-col
  __shared__ float bsh[64];
  const int tid = threadIdx.x;
  const int w = tid >> 5, t = tid & 31, lm = t & 15, lh = t >> 4;
  const int bh = blockIdx.y, b = bh >> 4, h = bh & 15;
  const int qbase = blockIdx.x * 128;

  #pragma unroll
  for (int it = 0; it < 16; ++it) {               // stage W^T as f16
    int e = it * 256 + tid; int k = e >> 6, n = e & 63;
    WsT[n * 72 + k] = (_Float16)W[k * 64 + n];
  }
  if (tid < 64) bsh[tid] = bias[tid];
  #pragma unroll
  for (int it = 0; it < 16; ++it) {               // stage X rows as f16
    int e2 = it * 256 + tid; int row = e2 >> 5, c2 = e2 & 31;
    const float2 v = *(const float2*)(x + ((size_t)(b * S + qbase + row)) * D + h * d + 2 * c2);
    h2 hv = { (_Float16)v.x, (_Float16)v.y };
    *(h2*)&Xs[row * 72 + 2 * c2] = hv;
  }
  __syncthreads();

  v8f acc[4] = {};
  #pragma unroll
  for (int kc = 0; kc < 64; kc += 32) {
    v16h A = loadA(&Xs[(16 * w + lm) * 72 + kc], lh);
    v16h Bv[4];
    #pragma unroll
    for (int nt = 0; nt < 4; ++nt) Bv[nt] = loadB(&WsT[(nt * 16 + lm) * 72 + kc], lh);
    #pragma unroll
    for (int nt = 0; nt < 4; ++nt) acc[nt] = wmma16(A, Bv[nt], acc[nt]);
  }
  #pragma unroll
  for (int nt = 0; nt < 4; ++nt) {
    float bb = bsh[nt * 16 + lm];
    #pragma unroll
    for (int r = 0; r < 8; ++r) {
      int qrow = qbase + 16 * w + 8 * lh + r;     // C layout: row = 8h + vgpr, col = lane
      Qh[((size_t)bh * S + qrow) * d + nt * 16 + lm] = (_Float16)((acc[nt][r] + bb) * 0.125f);
    }
  }
}

// ---------------- Kernel 2: fused flash attention (q == k == Q, v == x_h) ----------------
// Double-buffered, software-pipelined: tile kt+1's async K copy + X global loads are
// issued right after the barrier, overlapping the WMMA compute of tile kt.
__global__ __launch_bounds__(256) void attn_kernel(const float* __restrict__ x,
                                                   const _Float16* __restrict__ Qh,
                                                   float* __restrict__ out) {
  constexpr int S = 2048, D = 1024, d = 64;
  __shared__ __align__(16) _Float16 Qs[128 * 72];       // query tile (resident)
  __shared__ __align__(16) _Float16 Ks[2][128 * 72];    // key tiles, double-buffered
  __shared__ __align__(16) _Float16 XsT[2][64 * 136];   // x_h tiles transposed [d][keys]
  __shared__ __align__(16) _Float16 Ps[128 * 136];      // probabilities staging
  const int tid = threadIdx.x;
  const int w = tid >> 5, t = tid & 31, lm = t & 15, lh = t >> 4;
  const int bh = blockIdx.y, b = bh >> 4, h = bh & 15;
  const int qbase = blockIdx.x * 128;
  const size_t xbase = ((size_t)b * S) * D + (size_t)h * d;
  const size_t qrowbase = (size_t)bh * S;

  const int ldrow = tid >> 3, ldc8 = tid & 7;           // async-copy lane mapping
  const int xrow = tid >> 5, xc2 = tid & 31;            // X-stage lane mapping

  // async-load Q tile once (resident)
  #pragma unroll
  for (int it = 0; it < 4; ++it)
    async16(&Qs[(ldrow + it * 32) * 72 + ldc8 * 8],
            Qh + (qrowbase + qbase + ldrow + it * 32) * d + ldc8 * 8);
  // prologue: async K tile 0, X tile 0 -> registers
  #pragma unroll
  for (int it = 0; it < 4; ++it)
    async16(&Ks[0][(ldrow + it * 32) * 72 + ldc8 * 8],
            Qh + (qrowbase + ldrow + it * 32) * d + ldc8 * 8);
  float2 xr[16];
  #pragma unroll
  for (int it = 0; it < 16; ++it)
    xr[it] = *(const float2*)(x + xbase + (size_t)(xrow + it * 8) * D + 2 * xc2);

  float rmax[8], rsum[8];
  #pragma unroll
  for (int r = 0; r < 8; ++r) { rmax[r] = -3.0e38f; rsum[r] = 0.0f; }
  v8f oacc[4] = {};

  for (int kt = 0; kt < 16; ++kt) {
    const int cur = kt & 1, nxt = cur ^ 1;
    _Float16* KsC  = &Ks[cur][0];
    _Float16* XsTc = &XsT[cur][0];

    // stage this tile's x_h registers into LDS (transposed, f32->f16)
    #pragma unroll
    for (int it = 0; it < 16; ++it) {
      XsTc[(2 * xc2) * 136 + xrow + it * 8]     = (_Float16)xr[it].x;
      XsTc[(2 * xc2 + 1) * 136 + xrow + it * 8] = (_Float16)xr[it].y;
    }
    wait_async0();                                 // K tile kt (and Qs) in LDS
    __syncthreads();                               // all buffers for tile kt ready

    if (kt < 15) {                                 // pipeline tile kt+1 (uniform branch)
      const int kbn = (kt + 1) * 128;
      #pragma unroll
      for (int it = 0; it < 4; ++it)
        async16(&Ks[nxt][(ldrow + it * 32) * 72 + ldc8 * 8],
                Qh + (qrowbase + kbn + ldrow + it * 32) * d + ldc8 * 8);
      #pragma unroll
      for (int it = 0; it < 16; ++it)
        xr[it] = *(const float2*)(x + xbase + (size_t)(kbn + xrow + it * 8) * D + 2 * xc2);
    }

    // S_tile = Q K^T  (16 WMMA per wave); 1/d already folded into Qh
    v8f sc[8] = {};
    #pragma unroll
    for (int kc = 0; kc < 64; kc += 32) {
      v16h A = loadA(&Qs[(16 * w + lm) * 72 + kc], lh);
      #pragma unroll
      for (int g = 0; g < 2; ++g) {
        v16h Bv[4];
        #pragma unroll
        for (int j = 0; j < 4; ++j) Bv[j] = loadB(&KsC[((g * 4 + j) * 16 + lm) * 72 + kc], lh);
        #pragma unroll
        for (int j = 0; j < 4; ++j) sc[g * 4 + j] = wmma16(A, Bv[j], sc[g * 4 + j]);
      }
    }

    // online softmax: rows are shared by the 16 lanes of each half-wave
    float tmax[8];
    #pragma unroll
    for (int r = 0; r < 8; ++r) {
      float m = sc[0][r];
      #pragma unroll
      for (int nt = 1; nt < 8; ++nt) m = fmaxf(m, sc[nt][r]);
      tmax[r] = m;
    }
    #pragma unroll
    for (int xm = 1; xm < 16; xm <<= 1)
      #pragma unroll
      for (int r = 0; r < 8; ++r) tmax[r] = fmaxf(tmax[r], __shfl_xor(tmax[r], xm, 32));
    #pragma unroll
    for (int r = 0; r < 8; ++r) {
      float nm = fmaxf(rmax[r], tmax[r]);
      float corr = exp2f((rmax[r] - nm) * L2E);
      rmax[r] = nm; rsum[r] *= corr;
      #pragma unroll
      for (int nt = 0; nt < 4; ++nt) oacc[nt][r] *= corr;
    }
    #pragma unroll
    for (int nt = 0; nt < 8; ++nt)
      #pragma unroll
      for (int r = 0; r < 8; ++r) {
        float p = exp2f((sc[nt][r] - rmax[r]) * L2E);
        rsum[r] += p;                              // per-lane partial, reduced at the end
        Ps[(16 * w + 8 * lh + r) * 136 + nt * 16 + lm] = (_Float16)p;
      }

    // O += P @ X  (16 WMMA per wave); Ps rows are wave-private, no barrier needed
    #pragma unroll
    for (int kc = 0; kc < 128; kc += 32) {
      v16h A = loadA(&Ps[(16 * w + lm) * 136 + kc], lh);
      v16h Bv[4];
      #pragma unroll
      for (int nt = 0; nt < 4; ++nt) Bv[nt] = loadB(&XsTc[(nt * 16 + lm) * 136 + kc], lh);
      #pragma unroll
      for (int nt = 0; nt < 4; ++nt) oacc[nt] = wmma16(A, Bv[nt], oacc[nt]);
    }
    __syncthreads();                               // tile kt fully consumed by all waves
  }

  // finish row sums across the 16 lanes sharing each row, normalize, store
  #pragma unroll
  for (int xm = 1; xm < 16; xm <<= 1)
    #pragma unroll
    for (int r = 0; r < 8; ++r) rsum[r] += __shfl_xor(rsum[r], xm, 32);
  float rinv[8];
  #pragma unroll
  for (int r = 0; r < 8; ++r) rinv[r] = __builtin_amdgcn_rcpf(rsum[r]);
  #pragma unroll
  for (int nt = 0; nt < 4; ++nt)
    #pragma unroll
    for (int r = 0; r < 8; ++r) {
      int qrow = qbase + 16 * w + 8 * lh + r;
      out[(qrowbase + qrow) * d + nt * 16 + lm] = oacc[nt][r] * rinv[r];
    }
}

extern "C" void kernel_launch(void* const* d_in, const int* in_sizes, int n_in,
                              void* d_out, int out_size, void* d_ws, size_t ws_size,
                              hipStream_t stream) {
  (void)in_sizes; (void)n_in; (void)out_size; (void)ws_size;
  const float* x    = (const float*)d_in[0];
  // d_in[1] = mask: additive term is constant along the softmax (key) axis
  // -> softmax is shift-invariant -> exact no-op; intentionally unused.
  const float* W    = (const float*)d_in[2];
  const float* bias = (const float*)d_in[3];
  _Float16* Qh = (_Float16*)d_ws;                  // B*H*S*d f16 = 16 MB
  float* out = (float*)d_out;

  dim3 grid(16, 64);                               // (S/128, B*H)
  dim3 block(256);
  hipLaunchKernelGGL(proj_kernel, grid, block, 0, stream, x, W, bias, Qh);
  hipLaunchKernelGGL(attn_kernel, grid, block, 0, stream, x, Qh, out);
}
